// Distance_54030688584305
// MI455X (gfx1250) — matrix operand order repaired and checked by reading, not compiled, not run on hardware
//
#include <hip/hip_runtime.h>

// ---------------------------------------------------------------------------
// Pairwise squared Euclidean distance via GEMM identity on CDNA5 (gfx1250).
//   dist[n,q] = ||s_n||^2 + ||t_q||^2 - 2 * (s . t^T)[n,q]
//
// Pass 1 (prep, ~µs): split fp32 -> bf16 hi/lo ONCE into workspace, stored in
//   exact WMMA fragment layout (A-swizzle for s, B-swizzle for t + XOR bank
//   swizzle); plus exact fp32 row norms. ~8.1 MB -> L2-resident.
// Pass 2 (GEMM): 128x128 tile per 256-thread workgroup (8 wave32 waves,
//   16x128 per wave). B tile (32 KB hi+lo) staged once into LDS via a linear
//   copy (kills the 8x inter-wave L2 redundancy); fragments fed by
//   conflict-free ds_load_b128. A fragments are per-wave-unique coalesced
//   global_load_b128 from L2.  cross = a_hi*b_hi + a_hi*b_lo + a_lo*b_hi
//   (3x v_wmma_f32_16x16x32_bf16) ~ fp32 accuracy at bf16-WMMA rate.
//   Output (1 GiB) streamed with non-temporal stores = the only HBM traffic
//   (~46 us floor @ 23.3 TB/s), compute ~43 us fully overlapped.
// ---------------------------------------------------------------------------

typedef __attribute__((ext_vector_type(16))) __bf16 v16bf;
typedef __attribute__((ext_vector_type(8)))  float  v8f;

#define DIM   64        // K dimension (fixed by the problem)
#define NTILE 128       // output rows per workgroup
#define QTILE 128       // output cols per workgroup

union Frag {
    v16bf          v;
    unsigned short u[16];
    uint4          q4[2];
};

__device__ __forceinline__ unsigned short bf16_hi_trunc(float x) {
    union { float f; unsigned u; } c; c.f = x;
    return (unsigned short)(c.u >> 16);
}
__device__ __forceinline__ float bf16_to_f32(unsigned short h) {
    union { float f; unsigned u; } c; c.u = ((unsigned)h) << 16;
    return c.f;
}
__device__ __forceinline__ void split_bf16(float x, unsigned short& h, unsigned short& l) {
    h = bf16_hi_trunc(x);
    l = bf16_hi_trunc(x - bf16_to_f32(h));
}
__device__ __forceinline__ void split4(const float4 v, Frag& hi, Frag& lo, const int base) {
    split_bf16(v.x, hi.u[base + 0], lo.u[base + 0]);
    split_bf16(v.y, hi.u[base + 1], lo.u[base + 1]);
    split_bf16(v.z, hi.u[base + 2], lo.u[base + 2]);
    split_bf16(v.w, hi.u[base + 3], lo.u[base + 3]);
}

// ---------------------------------------------------------------------------
// Prep 1: fp32 -> bf16 hi/lo in WMMA fragment layout. One wave per 16x32 tile.
//   A (s) per-lane (lane = m + 16*h): u[0..7] = K h*8+{0..7}, u[8..15] = +16
//   B (t) per-lane (lane = m + 16*h): u[0..7] = K h*8+{0..7}, u[8..15] = +16
// Flat: frag[(tile*32 + lane)*16 + pos]. B planes additionally XOR-swizzle
// the two 16B chunks by (lane>>3)&1 so later ds_load_b128 is bank-conflict-
// free (self-inverse; loader applies the same XOR).
// ---------------------------------------------------------------------------
__global__ __launch_bounds__(256)
void prep_convert_kernel(const float* __restrict__ S, const float* __restrict__ T,
                         unsigned short* __restrict__ sA_hi, unsigned short* __restrict__ sA_lo,
                         unsigned short* __restrict__ tB_hi, unsigned short* __restrict__ tB_lo,
                         int N, int Q) {
    const int lane = threadIdx.x & 31;
    const int wv   = threadIdx.x >> 5;
    const int tile = blockIdx.x * 8 + wv;
    const int m    = lane & 15;
    const int h    = lane >> 4;
    const int nATiles = (N >> 4) * (DIM / 32);

    Frag hi, lo;
    if (tile < nATiles) {
        const int ri = tile >> 1, ki = tile & 1;
        const float* src = &S[(size_t)(ri * 16 + m) * DIM + ki * 32];
        split4(*(const float4*)&src[h * 8 + 0],      hi, lo, 0);
        split4(*(const float4*)&src[h * 8 + 4],      hi, lo, 4);
        split4(*(const float4*)&src[16 + h * 8 + 0], hi, lo, 8);
        split4(*(const float4*)&src[16 + h * 8 + 4], hi, lo, 12);
        const size_t base = ((size_t)tile * 32 + lane) * 16;
        *(uint4*)&sA_hi[base]     = hi.q4[0];
        *(uint4*)&sA_hi[base + 8] = hi.q4[1];
        *(uint4*)&sA_lo[base]     = lo.q4[0];
        *(uint4*)&sA_lo[base + 8] = lo.q4[1];
    } else {
        const int bt = tile - nATiles;
        const int qi = bt >> 1, ki = bt & 1;
        const float* src = &T[(size_t)(qi * 16 + m) * DIM + ki * 32 + h * 8];
        split4(*(const float4*)&src[0],  hi, lo, 0);
        split4(*(const float4*)&src[4],  hi, lo, 4);
        split4(*(const float4*)&src[16], hi, lo, 8);
        split4(*(const float4*)&src[20], hi, lo, 12);
        const int sw = (lane >> 3) & 1;                  // bank swizzle
        const size_t base = ((size_t)bt * 32 + lane) * 16;
        *(uint4*)&tB_hi[base + (0 ^ sw) * 8] = hi.q4[0];
        *(uint4*)&tB_hi[base + (1 ^ sw) * 8] = hi.q4[1];
        *(uint4*)&tB_lo[base + (0 ^ sw) * 8] = lo.q4[0];
        *(uint4*)&tB_lo[base + (1 ^ sw) * 8] = lo.q4[1];
    }
}

// ---------------------------------------------------------------------------
// Prep 2: exact fp32 row norms. One thread per row (s rows, then t rows).
// ---------------------------------------------------------------------------
__global__ __launch_bounds__(256)
void prep_norms_kernel(const float* __restrict__ S, const float* __restrict__ T,
                       float* __restrict__ ssq, float* __restrict__ tsq,
                       int N, int Q) {
    const int i = blockIdx.x * 256 + threadIdx.x;
    const float* src = (i < N) ? &S[(size_t)i * DIM] : &T[(size_t)(i - N) * DIM];
    float acc = 0.0f;
    #pragma unroll
    for (int k = 0; k < DIM; k += 4) {
        const float4 v = *(const float4*)&src[k];
        acc += v.x * v.x + v.y * v.y + v.z * v.z + v.w * v.w;
    }
    if (i < N) ssq[i] = acc;
    else       tsq[i - N] = acc;
}

// ---------------------------------------------------------------------------
// GEMM + epilogue. Workgroup = 128x128 output tile; wave wv = rows
// [wv*16, wv*16+16) x 128 cols (8 WMMA column tiles). B tile lives in LDS.
// ---------------------------------------------------------------------------
__global__ __launch_bounds__(256)
void Distance_54030688584305_kernel(const unsigned short* __restrict__ sA_hi,
                                    const unsigned short* __restrict__ sA_lo,
                                    const unsigned short* __restrict__ tB_hi,
                                    const unsigned short* __restrict__ tB_lo,
                                    const float* __restrict__ ssq,
                                    const float* __restrict__ tsq,
                                    float* __restrict__ out,
                                    int N, int Q) {
    // 8 j-tiles x 2 ki x 32 lanes x 16 ushorts = 8192 ushorts = 16 KB / plane
    __shared__ unsigned short lds_b_hi[(QTILE / 16) * 2 * 32 * 16];
    __shared__ unsigned short lds_b_lo[(QTILE / 16) * 2 * 32 * 16];

    const int tid   = threadIdx.x;
    const int lane  = tid & 31;
    const int wv    = tid >> 5;             // wave 0..7
    const int m     = lane & 15;
    const int kh    = lane >> 4;
    const int sw    = (lane >> 3) & 1;      // LDS bank swizzle (matches prep)
    const int nBase = blockIdx.y * NTILE;
    const int qBase = blockIdx.x * QTILE;
    const int riG   = (nBase >> 4) + wv;    // global A row-tile of this wave

    // --- stage B tile into LDS: linear copy (layout is wg-contiguous) -----
    {
        const size_t gb = (size_t)(qBase >> 4) * 2 * 32 * 16;  // ushort index
        const uint4* gH = (const uint4*)&tB_hi[gb];
        const uint4* gL = (const uint4*)&tB_lo[gb];
        uint4* lH = (uint4*)lds_b_hi;
        uint4* lL = (uint4*)lds_b_lo;
        #pragma unroll
        for (int c = 0; c < 4; ++c) {       // 1024 uint4 per plane / 256 thr
            lH[tid + 256 * c] = gH[tid + 256 * c];
            lL[tid + 256 * c] = gL[tid + 256 * c];
        }
    }
    __syncthreads();

    // --- WMMA main loop ----------------------------------------------------
    v8f acc[QTILE / 16] = {};               // 8 accumulators x 8 VGPRs

    #pragma unroll
    for (int ki = 0; ki < DIM / 32; ++ki) {
        Frag a_hi, a_lo;
        const size_t abase = (((size_t)riG * (DIM / 32) + ki) * 32 + lane) * 16;
        a_hi.q4[0] = *(const uint4*)&sA_hi[abase];
        a_hi.q4[1] = *(const uint4*)&sA_hi[abase + 8];
        a_lo.q4[0] = *(const uint4*)&sA_lo[abase];
        a_lo.q4[1] = *(const uint4*)&sA_lo[abase + 8];

        #pragma unroll
        for (int j = 0; j < QTILE / 16; ++j) {
            const int bbase = ((j * 2 + ki) * 32 + lane) * 16;
            Frag b_hi, b_lo;
            b_hi.q4[0] = *(const uint4*)&lds_b_hi[bbase + (0 ^ sw) * 8];
            b_hi.q4[1] = *(const uint4*)&lds_b_hi[bbase + (1 ^ sw) * 8];
            b_lo.q4[0] = *(const uint4*)&lds_b_lo[bbase + (0 ^ sw) * 8];
            b_lo.q4[1] = *(const uint4*)&lds_b_lo[bbase + (1 ^ sw) * 8];

            acc[j] = __builtin_amdgcn_wmma_f32_16x16x32_bf16(
                false, a_hi.v, false, b_hi.v, (short)0, acc[j], false, false);
            acc[j] = __builtin_amdgcn_wmma_f32_16x16x32_bf16(
                false, a_hi.v, false, b_lo.v, (short)0, acc[j], false, false);
            acc[j] = __builtin_amdgcn_wmma_f32_16x16x32_bf16(
                false, a_lo.v, false, b_hi.v, (short)0, acc[j], false, false);
        }
    }

    // epilogue: dist = ssq + tsq - 2*cross   (C/D: VGPR r -> M = r + 8*kh, N = m)
    float srow[8];
    {
        const float4* p = (const float4*)&ssq[nBase + wv * 16 + kh * 8];
        const float4 x = p[0], y = p[1];
        srow[0] = x.x; srow[1] = x.y; srow[2] = x.z; srow[3] = x.w;
        srow[4] = y.x; srow[5] = y.y; srow[6] = y.z; srow[7] = y.w;
    }
    #pragma unroll
    for (int j = 0; j < QTILE / 16; ++j) {
        const float tq = tsq[qBase + j * 16 + m];
        const size_t base =
            (size_t)(nBase + wv * 16 + kh * 8) * (size_t)Q + (qBase + j * 16 + m);
        #pragma unroll
        for (int r = 0; r < 8; ++r) {
            const float dv = srow[r] + tq - 2.0f * acc[j][r];
            __builtin_nontemporal_store(dv, &out[base + (size_t)r * Q]);
        }
    }
}

extern "C" void kernel_launch(void* const* d_in, const int* in_sizes, int n_in,
                              void* d_out, int out_size, void* d_ws, size_t ws_size,
                              hipStream_t stream) {
    const float* s = (const float*)d_in[0];
    const float* t = (const float*)d_in[1];
    float* out = (float*)d_out;
    const int N = in_sizes[0] / DIM;   // 16384
    const int Q = in_sizes[1] / DIM;   // 16384

    // workspace layout (bytes): 4 bf16 planes + 2 norm vectors  (~8.1 MB)
    const size_t szS = (size_t)N * DIM * sizeof(unsigned short);  // 2 MB
    const size_t szT = (size_t)Q * DIM * sizeof(unsigned short);  // 2 MB
    char* ws = (char*)d_ws;
    unsigned short* sA_hi = (unsigned short*)(ws);
    unsigned short* sA_lo = (unsigned short*)(ws + szS);
    unsigned short* tB_hi = (unsigned short*)(ws + 2 * szS);
    unsigned short* tB_lo = (unsigned short*)(ws + 2 * szS + szT);
    float*          ssq   = (float*)(ws + 2 * szS + 2 * szT);
    float*          tsq   = (float*)(ws + 2 * szS + 2 * szT + (size_t)N * sizeof(float));

    // Pass 1a: operand conversion into fragment layout (one wave per 16x32 tile)
    const int nTiles = (N >> 4) * (DIM / 32) + (Q >> 4) * (DIM / 32);  // 4096
    prep_convert_kernel<<<dim3(nTiles / 8), dim3(256), 0, stream>>>(
        s, t, sA_hi, sA_lo, tB_hi, tB_lo, N, Q);

    // Pass 1b: exact fp32 row norms (one thread per row)
    prep_norms_kernel<<<dim3((N + Q) / 256), dim3(256), 0, stream>>>(
        s, t, ssq, tsq, N, Q);

    // Pass 2: WMMA GEMM + distance epilogue
    dim3 grid(Q / QTILE, N / NTILE);   // (128, 128)
    dim3 block(256);                   // 8 wave32 waves
    Distance_54030688584305_kernel<<<grid, block, 0, stream>>>(
        sA_hi, sA_lo, tB_hi, tB_lo, ssq, tsq, out, N, Q);
}
